// EntityMapping_29068338659988
// MI455X (gfx1250) — compile-verified, matrix operand order checked by reading, not run on hardware
//
#include <hip/hip_runtime.h>
#include <hip/hip_bf16.h>

typedef __attribute__((ext_vector_type(16))) _Float16 v16h;
typedef __attribute__((ext_vector_type(8)))  _Float16 v8h;
typedef __attribute__((ext_vector_type(4)))  _Float16 v4h;
typedef __attribute__((ext_vector_type(8)))  float    v8f;

#define P_    16
#define E_    512
#define H_    512
#define N_    8192
#define ROWS_ 64

// ---------------------------------------------------------------------------
// Pre-pass: fp32 -> f16 convert (embedding), coalesced float4 -> 4x f16
// ---------------------------------------------------------------------------
__global__ __launch_bounds__(256) void k_cvt_f16(const float* __restrict__ in,
                                                 _Float16* __restrict__ out) {
    int i = (blockIdx.x * 256 + threadIdx.x) * 4;
    float4 v = *(const float4*)(in + i);
    v4h o = { (_Float16)v.x, (_Float16)v.y, (_Float16)v.z, (_Float16)v.w };
    *(v4h*)(out + i) = o;
}

// ---------------------------------------------------------------------------
// Pre-pass: [P,R,C] fp32 -> [P,C,R] f16 (LDS-tiled transpose + convert).
// Makes W K-contiguous per output column so B fragments load exactly like
// A fragments (ISA 7.12.2 16-bit operand layout).
// ---------------------------------------------------------------------------
__global__ __launch_bounds__(256) void k_transpose_cvt(const float* __restrict__ in,
                                                       _Float16* __restrict__ out,
                                                       int R, int C) {
    __shared__ float tile[32][33];
    const int pp = blockIdx.z;
    const int c0 = blockIdx.x * 32;
    const int r0 = blockIdx.y * 32;
    const float* src = in  + (size_t)pp * R * C;
    _Float16*    dst = out + (size_t)pp * R * C;
    const int tx = threadIdx.x, ty = threadIdx.y;
#pragma unroll
    for (int j = 0; j < 32; j += 8)
        tile[ty + j][tx] = src[(size_t)(r0 + ty + j) * C + c0 + tx];
    __syncthreads();
#pragma unroll
    for (int j = 0; j < 32; j += 8)
        dst[(size_t)(c0 + ty + j) * R + r0 + tx] = (_Float16)tile[tx][ty + j];
}

// ---------------------------------------------------------------------------
// Fragment helper: ISA 16-bit 16x32 A-operand layout.
//   lane l: row = l&15, K in {k0..k0+7} U {k0+16..k0+23}, k0 = (l>>4)*8
// Two 16B loads, fused into one v16h operand.
// ---------------------------------------------------------------------------
template <typename PT>
__device__ __forceinline__ v16h load_frag(const PT* base) {
    v8h lo = *(const v8h*)(base);
    v8h hi = *(const v8h*)(base + 16);
    return __builtin_shufflevector(lo, hi, 0, 1, 2, 3, 4, 5, 6, 7,
                                           8, 9, 10, 11, 12, 13, 14, 15);
}

// ---------------------------------------------------------------------------
// One fused MLP layer on a 64x512 tile:  sOut = relu(sIn[64,512] x Wt^T + b)
//   sIn  : LDS, row-major [64][512] f16
//   Wt   : global, [512 outCols][512 K] f16 (K contiguous)
//   sOut : LDS, row-major [64][512] f16
// Wave w owns output cols [64w, 64w+64). Register tile = 4 row-blocks x
// 4 col-blocks (16 accumulators): per K-step, 4 A frags (LDS) + 4 B frags
// (global) feed 16 WMMAs -> 0.5 ds_load + 0.5 global_load per WMMA.
// ---------------------------------------------------------------------------
__device__ __forceinline__ void gemm_layer(const _Float16* __restrict__ sIn,
                                           const _Float16* __restrict__ Wt,
                                           const float* __restrict__ bias,
                                           _Float16* __restrict__ sOut,
                                           int wave, int lane) {
    const int row   = lane & 15;
    const int k0    = (lane >> 4) << 3;
    const int hiOff = (lane >> 4) << 3;   // D-matrix: lanes 16-31 hold rows M+8

    const _Float16* wBase = Wt + (size_t)(wave * 64 + row) * 512;

    v8f acc[4][4];   // [cb][rb]
#pragma unroll
    for (int cb = 0; cb < 4; ++cb)
#pragma unroll
        for (int rb = 0; rb < 4; ++rb)
            acc[cb][rb] = (v8f){};

#pragma unroll 2
    for (int ks = 0; ks < 16; ++ks) {
        const int kb = ks * 32;
        v16h a[4], b[4];
#pragma unroll
        for (int rb = 0; rb < 4; ++rb)
            a[rb] = load_frag(sIn + (size_t)(rb * 16 + row) * 512 + kb + k0);
#pragma unroll
        for (int cb = 0; cb < 4; ++cb)
            b[cb] = load_frag(wBase + (size_t)cb * (16 * 512) + kb + k0);
#pragma unroll
        for (int cb = 0; cb < 4; ++cb)
#pragma unroll
            for (int rb = 0; rb < 4; ++rb)
                acc[cb][rb] = __builtin_amdgcn_wmma_f32_16x16x32_f16(
                    /*neg_a=*/false, a[rb], /*neg_b=*/false, b[cb],
                    /*c_mod=*/(short)0, acc[cb][rb],
                    /*reuse_a=*/false, /*reuse_b=*/false);
    }

    // bias + ReLU epilogue, write f16 back in next layer's A layout
#pragma unroll
    for (int cb = 0; cb < 4; ++cb) {
        const int n = wave * 64 + cb * 16 + row;
        const float bv = bias[n];
#pragma unroll
        for (int rb = 0; rb < 4; ++rb) {
#pragma unroll
            for (int i = 0; i < 8; ++i) {
                float v = acc[cb][rb][i] + bv;
                v = fmaxf(v, 0.0f);
                const int m = rb * 16 + i + hiOff;
                sOut[(size_t)m * 512 + n] = (_Float16)v;
            }
        }
    }
}

// ---------------------------------------------------------------------------
// Fused 3-layer MLP: one block = (partition p, 64-row tile of N)
// ---------------------------------------------------------------------------
__global__ __launch_bounds__(256) void k_mlp_fused(
    const _Float16* __restrict__ embH,  // [N,E] f16
    const _Float16* __restrict__ W0t,   // [P,H,E] f16 (K contiguous)
    const float* __restrict__ b0,       // [P,H]
    const _Float16* __restrict__ W1t,   // [P,H,H] f16 (K contiguous)
    const float* __restrict__ b1,       // [P,H]
    const float* __restrict__ W2,       // [P,H]
    const float* __restrict__ b2,       // [P]
    float* __restrict__ out)            // [N,P]
{
    __shared__ _Float16 sA[ROWS_ * E_];   // 64 KB: layer input
    __shared__ _Float16 sH[ROWS_ * H_];   // 64 KB: layer output
    __shared__ float    sRed[256];

    const int p     = blockIdx.x >> 7;        // 0..15 (slow-varying: L2 reuse)
    const int nTile = blockIdx.x & 127;       // 0..127
    const int nBase = nTile * ROWS_;
    const int tid   = threadIdx.x;
    const int wave  = tid >> 5;
    const int lane  = tid & 31;

    // ---- stage 64x512 embedding tile into LDS (16B coalesced) ----
    {
        const uint4* src = (const uint4*)(embH + (size_t)nBase * E_);
        uint4*       dst = (uint4*)sA;
#pragma unroll
        for (int i = 0; i < (ROWS_ * E_ / 8) / 256; ++i)   // 16 iters
            dst[tid + i * 256] = src[tid + i * 256];
    }
    __syncthreads();

    // ---- layer 0: sH = relu(sA x W0_p + b0) ----
    gemm_layer(sA, W0t + (size_t)p * H_ * E_, b0 + p * H_, sH, wave, lane);
    __syncthreads();

    // ---- layer 1: sA = relu(sH x W1_p + b1) ----
    gemm_layer(sH, W1t + (size_t)p * H_ * H_, b1 + p * H_, sA, wave, lane);
    __syncthreads();

    // ---- layer 2: out[n,p] = sigmoid(sA[n,:] . w2_p + b2_p) ----
    {
        const int r = tid >> 2;               // row 0..63
        const int q = tid & 3;                // quarter of K
        const v8h*    hv = (const v8h*)(sA + (size_t)r * H_ + q * 128);
        const float4* wv = (const float4*)(W2 + (size_t)p * H_ + q * 128);
        float s = 0.0f;
#pragma unroll 4
        for (int k = 0; k < 16; ++k) {
            v8h h8 = hv[k];
            float4 wa = wv[2 * k], wb = wv[2 * k + 1];
            s += (float)h8[0] * wa.x + (float)h8[1] * wa.y +
                 (float)h8[2] * wa.z + (float)h8[3] * wa.w;
            s += (float)h8[4] * wb.x + (float)h8[5] * wb.y +
                 (float)h8[6] * wb.z + (float)h8[7] * wb.w;
        }
        sRed[tid] = s;
    }
    __syncthreads();
    if (tid < ROWS_) {
        float s = sRed[tid * 4] + sRed[tid * 4 + 1] +
                  sRed[tid * 4 + 2] + sRed[tid * 4 + 3] + b2[p];
        out[(size_t)(nBase + tid) * P_ + p] = 1.0f / (1.0f + __expf(-s));
    }
}

// ---------------------------------------------------------------------------
// Host entry
// ---------------------------------------------------------------------------
extern "C" void kernel_launch(void* const* d_in, const int* in_sizes, int n_in,
                              void* d_out, int out_size, void* d_ws, size_t ws_size,
                              hipStream_t stream) {
    (void)in_sizes; (void)n_in; (void)out_size; (void)ws_size;
    const float* emb = (const float*)d_in[0];   // [N,E]
    const float* W0  = (const float*)d_in[1];   // [P,E,H]
    const float* b0  = (const float*)d_in[2];   // [P,H]
    const float* W1  = (const float*)d_in[3];   // [P,H,H]
    const float* b1  = (const float*)d_in[4];   // [P,H]
    const float* W2  = (const float*)d_in[5];   // [P,H,1]
    const float* b2  = (const float*)d_in[6];   // [P,1]
    float* out = (float*)d_out;

    char* ws = (char*)d_ws;
    _Float16* embH = (_Float16*)ws;                                   // N*E f16
    _Float16* W0t  = (_Float16*)(ws + (size_t)N_ * E_ * 2);           // P*H*E f16
    _Float16* W1t  = (_Float16*)(ws + (size_t)N_ * E_ * 2
                                    + (size_t)P_ * H_ * E_ * 2);      // P*H*H f16

    // pre-pass conversions (runs every call; ~25 MB traffic, L2-resident)
    k_cvt_f16<<<(N_ * E_ / 4) / 256, 256, 0, stream>>>(emb, embH);
    dim3 tb(32, 8);
    k_transpose_cvt<<<dim3(H_ / 32, E_ / 32, P_), tb, 0, stream>>>(W0, W0t, E_, H_);
    k_transpose_cvt<<<dim3(H_ / 32, H_ / 32, P_), tb, 0, stream>>>(W1, W1t, H_, H_);

    // fused MLP: 16 partitions x 128 row-tiles
    k_mlp_fused<<<P_ * (N_ / ROWS_), 256, 0, stream>>>(
        embH, W0t, b0, W1t, b1, W2, b2, out);
}
